// MoEModel_67843303408003
// MI455X (gfx1250) — matrix operand order repaired and checked by reading, not compiled, not run on hardware
//
#include <hip/hip_runtime.h>

typedef __bf16 bf16;
typedef __attribute__((ext_vector_type(16))) bf16  v16bf;
typedef __attribute__((ext_vector_type(8)))  bf16  v8bf;
typedef __attribute__((ext_vector_type(8)))  float v8f;

__device__ __forceinline__ bf16 f2bf(float f) {
  unsigned u = __builtin_bit_cast(unsigned, f);
  unsigned r = u + 0x7FFFu + ((u >> 16) & 1u);     // round-to-nearest-even
  unsigned short h = (unsigned short)(r >> 16);
  return __builtin_bit_cast(bf16, h);
}
__device__ __forceinline__ bf16 bfzero() {
  return __builtin_bit_cast(bf16, (unsigned short)0);
}

// ---------------------------------------------------------------------------
// Pure-bf16 tiled WMMA GEMM:
//   OUT[base+m, n] = act( A[row(m), :K] @ WT[n, :K]^T + bias[n] )
// WT is the pre-transposed bf16 weight ([N][K]) so both A and B staging are
// contiguous b128 global loads -> b128 LDS stores (no conversion VALU).
// Block: 256 threads (8 wave32). Tile M=128, N=128, K-step=32.
// Wave tile 32x64 = 2x4 v_wmma_f32_16x16x32_bf16 accumulators (8 WMMA/K-step).
// Register-pipelined: next K-tile global loads overlap current WMMA work.
// Grouped (expert) mode via counts/offsets; optional row gather via rowIdx.
// ---------------------------------------------------------------------------
template <bool RELU, bool OUTBF16>
__global__ __launch_bounds__(256) void wmma_gemm_kernel(
    const bf16* __restrict__ A, int lda,
    const bf16* __restrict__ WT, size_t wStride,   // per-group [Nw][Kdim]
    const float* __restrict__ bias, int biasStride,
    void* __restrict__ OUTv, int ldo, int Kdim,
    const int* __restrict__ counts, int constM,
    const int* __restrict__ offsets,
    const int* __restrict__ rowIdx) {
  const int g    = blockIdx.z;
  const int cnt  = counts ? counts[g] : constM;
  const int m0   = blockIdx.y * 128;
  if (m0 >= cnt) return;                      // block-uniform: EXEC stays full
  const int n0   = blockIdx.x * 128;
  const int base = offsets ? offsets[g] : 0;
  const bf16* WTg = WT + (size_t)g * wStride;
  const float* bg = bias + (size_t)g * biasStride;

  __shared__ __align__(16) bf16 lA[128][40];  // [m][k], 80B row stride (16B aligned)
  __shared__ __align__(16) bf16 lB[128][40];  // [n][k]

  const int tid  = threadIdx.x;
  const int lane = tid & 31;
  const int wv   = tid >> 5;                  // wave 0..7
  const int wm   = wv & 3;                    // M sub-tile: 32 rows
  const int wn   = wv >> 2;                   // N sub-tile: 64 cols
  const int lhi  = lane >> 4;                 // half-wave
  const int llo  = lane & 15;

  // Staging map (identical for A and B): 2 threads per row, 16 elems each.
  const int ra = tid >> 1;
  const int ka = (tid & 1) * 16;

  long arow = -1;
  if (m0 + ra < cnt) {
    int r = base + m0 + ra;
    arow = rowIdx ? rowIdx[r] : r;
  }
  const bf16* bptr = WTg + (size_t)(n0 + ra) * Kdim + ka;

  v8bf zv;
#pragma unroll
  for (int j = 0; j < 8; ++j) zv[j] = bfzero();

  v8bf a0, a1, b0, b1;
  auto loadTiles = [&](int k0) {
    if (arow >= 0) {
      const bf16* ap = A + (size_t)arow * lda + k0 + ka;
      a0 = *(const v8bf*)ap;
      a1 = *(const v8bf*)(ap + 8);
    } else {
      a0 = zv; a1 = zv;
    }
    b0 = *(const v8bf*)(bptr + k0);
    b1 = *(const v8bf*)(bptr + k0 + 8);
  };

  const v8f fz = {0.f, 0.f, 0.f, 0.f, 0.f, 0.f, 0.f, 0.f};
  v8f acc[2][4];
#pragma unroll
  for (int i = 0; i < 2; ++i)
#pragma unroll
    for (int j = 0; j < 4; ++j) acc[i][j] = fz;

  loadTiles(0);
  for (int k0 = 0; k0 < Kdim; k0 += 32) {
    *(v8bf*)&lA[ra][ka]     = a0;
    *(v8bf*)&lA[ra][ka + 8] = a1;
    *(v8bf*)&lB[ra][ka]     = b0;
    *(v8bf*)&lB[ra][ka + 8] = b1;
    __syncthreads();

    if (k0 + 32 < Kdim) loadTiles(k0 + 32);   // overlap VMEM with WMMA below

    // A frag 16x32: lane M=llo; elems 0..7 -> K=8*lhi.., 8..15 -> K=16+8*lhi..
    v16bf af[2];
#pragma unroll
    for (int i = 0; i < 2; ++i) {
      const bf16* ap = &lA[wm * 32 + i * 16 + llo][lhi * 8];
      v8bf lo = *(const v8bf*)ap;
      v8bf hi = *(const v8bf*)(ap + 16);
#pragma unroll
      for (int j = 0; j < 8; ++j) { af[i][j] = lo[j]; af[i][j + 8] = hi[j]; }
    }
    // B frag 32x16: lane N=llo; elems 0..7 -> K=8*lhi.., 8..15 -> K=16+8*lhi..
    v16bf bfr[4];
#pragma unroll
    for (int i = 0; i < 4; ++i) {
      const bf16* bp = &lB[wn * 64 + i * 16 + llo][lhi * 8];
      v8bf lo = *(const v8bf*)bp;
      v8bf hi = *(const v8bf*)(bp + 16);
#pragma unroll
      for (int j = 0; j < 8; ++j) { bfr[i][j] = lo[j]; bfr[i][j + 8] = hi[j]; }
    }
#pragma unroll
    for (int i = 0; i < 2; ++i)
#pragma unroll
      for (int j = 0; j < 4; ++j)
        acc[i][j] = __builtin_amdgcn_wmma_f32_16x16x32_bf16(
            false, af[i], false, bfr[j], (short)0, acc[i][j], false, false);
    __syncthreads();
  }

  // ---- epilogue: bias, optional ReLU, fp32 or bf16 store ----
  const int mbase = wm * 32;
  const int nbase = n0 + wn * 64;
#pragma unroll
  for (int i = 0; i < 2; ++i) {
#pragma unroll
    for (int j = 0; j < 4; ++j) {
      const int nn = nbase + j * 16 + llo;
      const float bval = bg[nn];
#pragma unroll
      for (int r = 0; r < 8; ++r) {
        const int mloc = m0 + mbase + i * 16 + lhi * 8 + r;  // C: M = r + 8*halfwave
        if (mloc < cnt) {
          float v = acc[i][j][r] + bval;
          if (RELU) v = fmaxf(v, 0.f);
          size_t off = (size_t)(base + mloc) * ldo + nn;
          if constexpr (OUTBF16) ((bf16*)OUTv)[off] = f2bf(v);
          else                   ((float*)OUTv)[off] = v;
        }
      }
    }
  }
}

// ---------------------------------------------------------------------------
// Transpose + fp32->bf16 convert:  out[n][k] = bf16(in[k][n]) for each matrix.
// LDS-tiled so both global reads and writes are coalesced. Dims % 64 == 0.
// ---------------------------------------------------------------------------
__global__ __launch_bounds__(256) void transpose_cvt_kernel(
    const float* __restrict__ in, bf16* __restrict__ out, int Kd, int Nd) {
  const size_t mstride = (size_t)Kd * Nd;
  const float* ip = in + (size_t)blockIdx.z * mstride;
  bf16* op = out + (size_t)blockIdx.z * mstride;
  __shared__ bf16 tile[64][72];
  const int k0 = blockIdx.y * 64, n0 = blockIdx.x * 64;
  const int c  = threadIdx.x & 63;
  const int r0 = (threadIdx.x >> 6) * 16;
#pragma unroll
  for (int rr = 0; rr < 16; ++rr) {
    const int r = r0 + rr;                       // k within tile
    tile[c][r] = f2bf(ip[(size_t)(k0 + r) * Nd + (n0 + c)]);
  }
  __syncthreads();
#pragma unroll
  for (int rr = 0; rr < 16; ++rr) {
    const int n = r0 + rr;                       // n within tile
    op[(size_t)(n0 + n) * Kd + (k0 + c)] = tile[n][c];
  }
}

// ---------------------------------------------------------------------------
// Elementwise fp32 -> bf16
// ---------------------------------------------------------------------------
__global__ __launch_bounds__(256) void cvt_kernel(const float* __restrict__ in,
                                                  bf16* __restrict__ out, int n) {
  const int i = blockIdx.x * 256 + threadIdx.x;
  if (i < n) out[i] = f2bf(in[i]);
}

// ---------------------------------------------------------------------------
// LayerNorm rows of 768; optionally emits a bf16 copy for WMMA consumption.
// ---------------------------------------------------------------------------
__global__ __launch_bounds__(256) void ln_kernel(const float* __restrict__ in,
                                                 const float* __restrict__ g,
                                                 const float* __restrict__ b,
                                                 float* __restrict__ out,
                                                 bf16* __restrict__ outb) {
  const int row = blockIdx.x;
  const int t = threadIdx.x;
  const float* x = in + (size_t)row * 768;
  float v0 = x[t], v1 = x[t + 256], v2 = x[t + 512];
  __shared__ float rs[256], rq[256];
  rs[t] = v0 + v1 + v2;
  rq[t] = v0 * v0 + v1 * v1 + v2 * v2;
  __syncthreads();
  for (int o = 128; o > 0; o >>= 1) {
    if (t < o) { rs[t] += rs[t + o]; rq[t] += rq[t + o]; }
    __syncthreads();
  }
  const float mean = rs[0] * (1.f / 768.f);
  const float var  = rq[0] * (1.f / 768.f) - mean * mean;
  const float inv  = rsqrtf(var + 1e-5f);
  const float r0v = (v0 - mean) * inv * g[t]       + b[t];
  const float r1v = (v1 - mean) * inv * g[t + 256] + b[t + 256];
  const float r2v = (v2 - mean) * inv * g[t + 512] + b[t + 512];
  float* o_ = out + (size_t)row * 768;
  o_[t] = r0v; o_[t + 256] = r1v; o_[t + 512] = r2v;
  if (outb) {
    bf16* ob = outb + (size_t)row * 768;
    ob[t] = f2bf(r0v); ob[t + 256] = f2bf(r1v); ob[t + 512] = f2bf(r2v);
  }
}

// ---------------------------------------------------------------------------
// Gating: one wave per token -> 8 logits, top-2 (earliest index on ties, like
// lax.top_k), softmax over the two values.
// ---------------------------------------------------------------------------
__global__ __launch_bounds__(256) void gate_kernel(const float* __restrict__ z,
                                                   const float* __restrict__ wg,
                                                   int* __restrict__ top_idx,
                                                   float* __restrict__ top_w) {
  const int n = blockIdx.x * 8 + (threadIdx.x >> 5);
  const int lane = threadIdx.x & 31;
  const float* zr = z + (size_t)n * 768;
  float acc[8] = {0.f, 0.f, 0.f, 0.f, 0.f, 0.f, 0.f, 0.f};
  for (int d = lane; d < 768; d += 32) {
    const float zvv = zr[d];
    const float* wr = wg + (size_t)d * 8;
#pragma unroll
    for (int e = 0; e < 8; ++e) acc[e] += zvv * wr[e];
  }
#pragma unroll
  for (int off = 16; off > 0; off >>= 1)
#pragma unroll
    for (int e = 0; e < 8; ++e) acc[e] += __shfl_xor(acc[e], off, 32);
  if (lane == 0) {
    float b0 = -1e30f, b1 = -1e30f; int i0 = 0, i1 = 0;
#pragma unroll
    for (int e = 0; e < 8; ++e) {
      const float v = acc[e];
      if (v > b0)      { b1 = b0; i1 = i0; b0 = v; i0 = e; }
      else if (v > b1) { b1 = v; i1 = e; }
    }
    const float tt = expf(b1 - b0);
    const float w0 = 1.f / (1.f + tt);
    top_idx[2 * n] = i0; top_idx[2 * n + 1] = i1;
    top_w[2 * n] = w0;   top_w[2 * n + 1] = tt * w0;
  }
}

// ---------------------------------------------------------------------------
// Deterministic routing (ordered scan, no atomics).
// ---------------------------------------------------------------------------
__global__ void route_kernel(const int* __restrict__ top_idx,
                             int* __restrict__ counts, int* __restrict__ offsets,
                             int* __restrict__ asgn_token, int* __restrict__ slotOf) {
  const int t = threadIdx.x;
  __shared__ int scount[8];
  __shared__ int soff[9];
  if (t < 8) {
    int c = 0;
    for (int n = 0; n < 2048; ++n) {
      const int a = top_idx[2 * n], b = top_idx[2 * n + 1];
      if (a == t || b == t) ++c;
    }
    scount[t] = c; counts[t] = c;
  }
  __syncthreads();
  if (t == 0) {
    int o = 0;
    for (int e = 0; e < 8; ++e) { soff[e] = o; offsets[e] = o; o += scount[e]; }
    soff[8] = o; offsets[8] = o;
  }
  __syncthreads();
  if (t < 8) {
    int pos = soff[t];
    for (int n = 0; n < 2048; ++n) {
      const int a = top_idx[2 * n], b = top_idx[2 * n + 1];
      if (a == t)      { asgn_token[pos] = n; slotOf[2 * n]     = pos; ++pos; }
      else if (b == t) { asgn_token[pos] = n; slotOf[2 * n + 1] = pos; ++pos; }
    }
  }
}

// ---------------------------------------------------------------------------
// Combine: h += g0*y[s0] + g1*y[s1]; also keeps a bf16 mirror of h for the
// final WMMA projection. Fixed order -> deterministic.
// ---------------------------------------------------------------------------
__global__ __launch_bounds__(256) void combine_kernel(float* __restrict__ h,
                                                      bf16* __restrict__ h_bf,
                                                      const float* __restrict__ y,
                                                      const int* __restrict__ slotOf,
                                                      const float* __restrict__ top_w) {
  const int idx = blockIdx.x * 256 + threadIdx.x;
  const int n = idx / 768, d = idx - n * 768;
  const int s0 = slotOf[2 * n], s1 = slotOf[2 * n + 1];
  const float g0 = top_w[2 * n], g1 = top_w[2 * n + 1];
  const float v = h[idx] + g0 * y[(size_t)s0 * 768 + d] + g1 * y[(size_t)s1 * 768 + d];
  h[idx] = v;
  h_bf[idx] = f2bf(v);
}

// ---------------------------------------------------------------------------
extern "C" void kernel_launch(void* const* d_in, const int* in_sizes, int n_in,
                              void* d_out, int out_size, void* d_ws, size_t ws_size,
                              hipStream_t stream) {
  (void)in_sizes; (void)n_in; (void)out_size; (void)ws_size;
  constexpr int N = 2048, D = 768, E = 8, F = 3072, L = 4;

  const float* x        = (const float*)d_in[0];
  const float* w_in     = (const float*)d_in[1];
  const float* b_in     = (const float*)d_in[2];
  const float* g_in     = (const float*)d_in[3];
  const float* beta_in  = (const float*)d_in[4];
  const float* g_ln     = (const float*)d_in[5];
  const float* beta_ln  = (const float*)d_in[6];
  const float* w_gate   = (const float*)d_in[7];
  const float* w1       = (const float*)d_in[8];
  const float* b1       = (const float*)d_in[9];
  const float* w2       = (const float*)d_in[10];
  const float* b2       = (const float*)d_in[11];
  const float* w_out    = (const float*)d_in[12];
  const float* b_out    = (const float*)d_in[13];
  const float* g_out    = (const float*)d_in[14];
  const float* beta_out = (const float*)d_in[15];
  float* out = (float*)d_out;

  char* p = (char*)d_ws;
  auto take = [&](size_t bytes) { char* q = p; p += (bytes + 255) & ~(size_t)255; return q; };
  float* h       = (float*)take((size_t)N * D * 4);
  bf16*  h_bf    = (bf16*) take((size_t)N * D * 2);
  float* z       = (float*)take((size_t)N * D * 4);
  bf16*  z_bf    = (bf16*) take((size_t)N * D * 2);
  bf16*  x_bf    = (bf16*) take((size_t)N * D * 2);
  bf16*  winT    = (bf16*) take((size_t)D * D * 2);
  bf16*  woutT   = (bf16*) take((size_t)D * D * 2);
  bf16*  w1T     = (bf16*) take((size_t)E * F * D * 2);  // per-layer, [e][F][D]
  bf16*  w2T     = (bf16*) take((size_t)E * D * F * 2);  // per-layer, [e][D][F]
  bf16*  h1      = (bf16*) take((size_t)N * 2 * F * 2);
  float* y       = (float*)take((size_t)N * 2 * D * 4);
  int*   top_idx = (int*)  take((size_t)N * 2 * 4);
  float* top_w   = (float*)take((size_t)N * 2 * 4);
  int*   counts  = (int*)  take(64);
  int*   offsets = (int*)  take(64);
  int*   asgn    = (int*)  take((size_t)N * 2 * 4);
  int*   slotOf  = (int*)  take((size_t)N * 2 * 4);

  const dim3 blk(256);
  const int gy = (N + 127) / 128;   // 16

  // one-time conversions (graph-safe, deterministic)
  cvt_kernel<<<(N * D) / 256, blk, 0, stream>>>(x, x_bf, N * D);
  transpose_cvt_kernel<<<dim3(D / 64, D / 64, 1), blk, 0, stream>>>(w_in, winT, D, D);
  transpose_cvt_kernel<<<dim3(D / 64, D / 64, 1), blk, 0, stream>>>(w_out, woutT, D, D);

  // input projection: z = x @ w_in + b_in ; h = LN(z)
  wmma_gemm_kernel<false, false><<<dim3(D / 128, gy, 1), blk, 0, stream>>>(
      x_bf, D, winT, 0, b_in, 0, z, D, D, nullptr, N, nullptr, nullptr);
  ln_kernel<<<N, blk, 0, stream>>>(z, g_in, beta_in, h, nullptr);

  for (int l = 0; l < L; ++l) {
    ln_kernel<<<N, blk, 0, stream>>>(h, g_ln + (size_t)l * D, beta_ln + (size_t)l * D, z, z_bf);
    gate_kernel<<<N / 8, blk, 0, stream>>>(z, w_gate + (size_t)l * D * E, top_idx, top_w);
    route_kernel<<<1, blk, 0, stream>>>(top_idx, counts, offsets, asgn, slotOf);
    // convert this layer's expert weights to transposed bf16 (L2-resident)
    transpose_cvt_kernel<<<dim3(F / 64, D / 64, E), blk, 0, stream>>>(
        w1 + (size_t)l * E * D * F, w1T, D, F);
    transpose_cvt_kernel<<<dim3(D / 64, F / 64, E), blk, 0, stream>>>(
        w2 + (size_t)l * E * F * D, w2T, F, D);
    // FFN1: h1[assign, F] = relu( z[token] @ w1[l,e] + b1[l,e] )   (bf16 out)
    wmma_gemm_kernel<true, true><<<dim3(F / 128, gy, E), blk, 0, stream>>>(
        z_bf, D, w1T, (size_t)F * D, b1 + (size_t)l * E * F, F,
        h1, F, D, counts, 0, offsets, asgn);
    // FFN2: y[assign, D] = h1[assign] @ w2[l,e] + b2[l,e]          (fp32 out)
    wmma_gemm_kernel<false, false><<<dim3(D / 128, gy, E), blk, 0, stream>>>(
        h1, F, w2T, (size_t)D * F, b2 + (size_t)l * E * D, D,
        y, D, F, counts, 0, offsets, nullptr);
    combine_kernel<<<(N * D) / 256, blk, 0, stream>>>(h, h_bf, y, slotOf, top_w);
  }

  // output projection + final LN -> d_out
  wmma_gemm_kernel<false, false><<<dim3(D / 128, gy, 1), blk, 0, stream>>>(
      h_bf, D, woutT, 0, b_out, 0, z, D, D, nullptr, N, nullptr, nullptr);
  ln_kernel<<<N, blk, 0, stream>>>(z, g_out, beta_out, out, nullptr);
}
